// LstmModel_69372311765515
// MI455X (gfx1250) — matrix operand order repaired and checked by reading, not compiled, not run on hardware
//
#include <hip/hip_runtime.h>
#include <hip/hip_bf16.h>
#include <cstdint>

typedef __bf16 bf16;
typedef __bf16 bf16x8  __attribute__((ext_vector_type(8)));
typedef __bf16 bf16x16 __attribute__((ext_vector_type(16)));
typedef float  f32x4   __attribute__((ext_vector_type(4)));
typedef float  f32x8   __attribute__((ext_vector_type(8)));
typedef int    i32x4   __attribute__((ext_vector_type(4)));

constexpr int kH   = 512;   // hidden
constexpr int kB   = 512;   // batch
constexpr int kT   = 512;   // timesteps
constexpr int kTGT = 28;    // output targets
constexpr int kZ   = 256;   // fc1 width (H/2)

#if defined(__has_builtin)
#if __has_builtin(__builtin_amdgcn_global_load_async_to_lds_b128)
#define HAVE_ASYNC_LDS 1
#endif
#if __has_builtin(__builtin_amdgcn_s_wait_asynccnt)
#define HAVE_WAIT_ASYNC 1
#endif
#endif

#if defined(HAVE_ASYNC_LDS)
typedef __attribute__((address_space(1))) i32x4 gbl_i32x4;  // global int4
typedef __attribute__((address_space(3))) i32x4 lds_i32x4;  // LDS int4
#endif

// ---------------------------------------------------------------------------
// Pack W_hh [2048,512] f32 into bf16 WMMA B-operand blocks.
// Block index = n_tile*16 + kk; block = 32 lanes x 16 bf16 lane-contiguous,
// element order v0.lo, v0.hi, ... v7.hi per CDNA5 16-bit B layout:
//   lanes 0-15:  N = lane,    K = j (j<8), 16 + (j-8) (j>=8)
//   lanes 16-31: N = lane-16, K = 8 + j (j<8), 24 + (j-8) (j>=8)
// ---------------------------------------------------------------------------
__global__ __launch_bounds__(256) void pack_whh(const float* __restrict__ Whh,
                                                bf16* __restrict__ Wp) {
  int idx    = blockIdx.x * 256 + threadIdx.x;   // 0 .. 2048*512-1
  int blk    = idx >> 9;                         // 512 elems per block
  int within = idx & 511;
  int lane   = within >> 4;
  int j      = within & 15;                      // v = j>>1, hi/lo = j&1
  int n_tile = blk >> 4;
  int kk     = blk & 15;
  int n      = lane & 15;
  int klocal = ((lane < 16) ? 0 : 8) + (j & 7) + ((j >> 3) << 4);
  int grow   = n_tile * 16 + n;                  // gate (row of W_hh)
  int gcol   = kk * 32 + klocal;                 // k   (col of W_hh)
  Wp[idx] = (bf16)Whh[grow * kH + gcol];
}

__global__ __launch_bounds__(256) void init_state(bf16* __restrict__ h0,
                                                  float* __restrict__ c0) {
  int idx = blockIdx.x * 256 + threadIdx.x;      // 512*512 elements
  h0[idx] = (bf16)0.0f;
  c0[idx] = 0.0f;
}

// ---------------------------------------------------------------------------
// One LSTM timestep. Grid: (8 colgroups, 16 rowpairs), block = 128 (4 waves).
// Wave w owns h-col tile hc = blockIdx.x*4 + w and a 32-row batch slab:
// 2 row-tiles x 4 gate tiles = 8 f32 accumulators, K = 512 reduction via
// 8 x v_wmma_f32_16x16x32_bf16 per K-slice (16 slices = 128 WMMA/wave).
// All B addresses are a single base + constant immediate offsets.
// ---------------------------------------------------------------------------
__global__ __launch_bounds__(128) void lstm_step(
    const bf16*  __restrict__ Wp,
    const bf16*  __restrict__ h_in,
    bf16*        __restrict__ h_out,
    float*       __restrict__ c_st,
    float*       __restrict__ h_f32,
    const float* __restrict__ seq,     // [B, T, 1]
    const float* __restrict__ W_ih,    // [4H, 1]
    const float* __restrict__ b_ih,    // [4H]
    const float* __restrict__ b_hh,    // [4H]
    int t) {
  __shared__ __align__(16) bf16 Ash[32 * kH];    // 32 rows x 512 cols = 32 KB

  const int rowpair = blockIdx.y;                // batch rows rowpair*32..+32
  const int tid     = threadIdx.x;
  const int wave    = tid >> 5;
  const int lane    = tid & 31;
  const int hc      = blockIdx.x * 4 + wave;     // h-col tile 0..31

  // ---- stage A operand (h rows rowpair*32..+32, all K) into LDS ----
  const char* src = (const char*)(h_in + rowpair * 32 * kH);  // 32 KB region
#if defined(HAVE_ASYNC_LDS)
  {
    auto* ldst = (__attribute__((address_space(3))) char*)Ash;
    auto* gsrc = (__attribute__((address_space(1))) char*)src;
#pragma unroll
    for (int i = 0; i < 16; ++i) {
      int off = (i * 128 + tid) * 16;
      __builtin_amdgcn_global_load_async_to_lds_b128(
          (gbl_i32x4*)(gsrc + off), (lds_i32x4*)(ldst + off), 0, 0);
    }
#if defined(HAVE_WAIT_ASYNC)
    __builtin_amdgcn_s_wait_asynccnt(0);
#else
    asm volatile("s_wait_asynccnt 0" ::: "memory");
#endif
  }
#else
#pragma unroll
  for (int i = 0; i < 16; ++i) {
    int off = (i * 128 + tid) * 16;
    *(f32x4*)((char*)Ash + off) = *(const f32x4*)(src + off);
  }
#endif
  __syncthreads();

  // ---- K-loop: 16 slices of 32; 2 row-tiles x 4 gates per slice ----
  f32x8 acc[8] = {f32x8{}, f32x8{}, f32x8{}, f32x8{},
                  f32x8{}, f32x8{}, f32x8{}, f32x8{}};
  const int m16   = lane & 15;
  const int kbase = (lane < 16) ? 0 : 8;   // CDNA5 16-bit A layout lane split

  // Constant-offset bases (element units; offsets become load immediates).
  const bf16* abase = &Ash[m16 * kH + kbase];       // + rt*8192 + kk*32 (+16)
  const bf16* bbase = Wp + hc * 8192 + lane * 16;   // + g*262144 + kk*512 (+8)

#pragma unroll
  for (int kk = 0; kk < 16; ++kk) {
    bf16x8 a0lo = *(const bf16x8*)(abase + kk * 32);
    bf16x8 a0hi = *(const bf16x8*)(abase + kk * 32 + 16);
    bf16x8 a1lo = *(const bf16x8*)(abase + 8192 + kk * 32);
    bf16x8 a1hi = *(const bf16x8*)(abase + 8192 + kk * 32 + 16);
    bf16x16 a0 = __builtin_shufflevector(a0lo, a0hi, 0, 1, 2, 3, 4, 5, 6, 7, 8,
                                         9, 10, 11, 12, 13, 14, 15);
    bf16x16 a1 = __builtin_shufflevector(a1lo, a1hi, 0, 1, 2, 3, 4, 5, 6, 7, 8,
                                         9, 10, 11, 12, 13, 14, 15);
#pragma unroll
    for (int g = 0; g < 4; ++g) {
      const bf16* bp = bbase + g * 262144 + kk * 512;
      bf16x8 blo = *(const bf16x8*)bp;
      bf16x8 bhi = *(const bf16x8*)(bp + 8);
      bf16x16 b  = __builtin_shufflevector(blo, bhi, 0, 1, 2, 3, 4, 5, 6, 7, 8,
                                           9, 10, 11, 12, 13, 14, 15);
      acc[g] = __builtin_amdgcn_wmma_f32_16x16x32_bf16(
          false, a0, false, b, (short)0, acc[g], false, false);
      acc[4 + g] = __builtin_amdgcn_wmma_f32_16x16x32_bf16(
          false, a1, false, b, (short)0, acc[4 + g], false, false);
    }
  }

  // ---- element-wise cell update for the two 16x16 tiles ----
  const int nl = lane & 15;
  const int n  = hc * 16 + nl;                       // hidden index
  float wih[4], bs[4];
#pragma unroll
  for (int g = 0; g < 4; ++g) {
    int gi = g * kH + n;
    wih[g] = W_ih[gi];
    bs[g]  = b_ih[gi] + b_hh[gi];
  }
#pragma unroll
  for (int r = 0; r < 8; ++r) {
#pragma unroll
    for (int rt = 0; rt < 2; ++rt) {
      // C/D layout: VGPR r -> M = r (lanes 0-15) / r+8 (lanes 16-31)
      int m = rowpair * 32 + rt * 16 + ((lane < 16) ? r : r + 8);  // batch
      float sv = seq[m * kT + t];
      float xi = acc[rt * 4 + 0][r] + sv * wih[0] + bs[0];
      float xf = acc[rt * 4 + 1][r] + sv * wih[1] + bs[1];
      float xg = acc[rt * 4 + 2][r] + sv * wih[2] + bs[2];
      float xo = acc[rt * 4 + 3][r] + sv * wih[3] + bs[3];
      float ig = 1.0f / (1.0f + __expf(-xi));
      float fg = 1.0f / (1.0f + __expf(-xf));
      float gg = tanhf(xg);
      float og = 1.0f / (1.0f + __expf(-xo));
      float cn = fg * c_st[m * kH + n] + ig * gg;
      c_st[m * kH + n] = cn;
      float hn = og * tanhf(cn);
      h_out[m * kH + n] = (bf16)hn;
      h_f32[m * kH + n] = hn;
    }
  }
}

// ---------------------------------------------------------------------------
// MLP head: z = relu(h @ fc1_w^T + fc1_b); out = z @ fc2_w^T + fc2_b.
// One block per batch row; 0.14 GFLOP total -> plain VALU is fine.
// ---------------------------------------------------------------------------
__global__ __launch_bounds__(256) void head_kernel(
    const float* __restrict__ hf, const float* __restrict__ fc1w,
    const float* __restrict__ fc1b, const float* __restrict__ fc2w,
    const float* __restrict__ fc2b, float* __restrict__ out) {
  __shared__ float z[kZ];
  int b = blockIdx.x;
  int j = threadIdx.x;
  const float* hr = hf + b * kH;
  const float* wr = fc1w + j * kH;
  float s = fc1b[j];
#pragma unroll 4
  for (int k = 0; k < kH; ++k) s += hr[k] * wr[k];
  z[j] = fmaxf(s, 0.0f);
  __syncthreads();
  if (j < kTGT) {
    const float* w2 = fc2w + j * kZ;
    float s2 = fc2b[j];
#pragma unroll 4
    for (int k = 0; k < kZ; ++k) s2 += z[k] * w2[k];
    out[b * kTGT + j] = s2;
  }
}

extern "C" void kernel_launch(void* const* d_in, const int* in_sizes, int n_in,
                              void* d_out, int out_size, void* d_ws,
                              size_t ws_size, hipStream_t stream) {
  const float* seq  = (const float*)d_in[0];
  const float* Wih  = (const float*)d_in[1];
  const float* Whh  = (const float*)d_in[2];
  const float* bih  = (const float*)d_in[3];
  const float* bhh  = (const float*)d_in[4];
  const float* fc1w = (const float*)d_in[5];
  const float* fc1b = (const float*)d_in[6];
  const float* fc2w = (const float*)d_in[7];
  const float* fc2b = (const float*)d_in[8];
  float* out = (float*)d_out;

  char* ws = (char*)d_ws;
  bf16* Wp = (bf16*)ws;    ws += (size_t)2048 * 512 * sizeof(bf16);  // 2 MB
  bf16* hA = (bf16*)ws;    ws += (size_t)512 * 512 * sizeof(bf16);   // 512 KB
  bf16* hB = (bf16*)ws;    ws += (size_t)512 * 512 * sizeof(bf16);   // 512 KB
  float* cst = (float*)ws; ws += (size_t)512 * 512 * sizeof(float);  // 1 MB
  float* hf = (float*)ws;  ws += (size_t)512 * 512 * sizeof(float);  // 1 MB

  pack_whh<<<4096, 256, 0, stream>>>(Whh, Wp);
  init_state<<<1024, 256, 0, stream>>>(hA, cst);

  dim3 grid(8, 16);  // 8 col-groups (4 waves each) x 16 row-pairs (32 rows)
  for (int t = 0; t < kT; ++t) {
    const bf16* hin = (t & 1) ? hB : hA;
    bf16* hout      = (t & 1) ? hA : hB;
    lstm_step<<<grid, 128, 0, stream>>>(Wp, hin, hout, cst, hf, seq, Wih, bih,
                                        bhh, t);
  }
  head_kernel<<<512, 256, 0, stream>>>(hf, fc1w, fc1b, fc2w, fc2b, out);
}
